// GCNTox21_83829171683414
// MI455X (gfx1250) — matrix operand order repaired and checked by reading, not compiled, not run on hardware
//
#include <hip/hip_runtime.h>
#include <hip/hip_bf16.h>

typedef _Float16 half_t;
typedef __attribute__((ext_vector_type(16))) _Float16 v16h;
typedef __attribute__((ext_vector_type(8)))  _Float16 v8h;
typedef __attribute__((ext_vector_type(8)))  float    v8f;

#define N_NODES 20000
#define N_EDGES 200000
#define F_NODE 32
#define F_EDGE 8
#define HDIM 256
#define EHDIM 16
#define GDIM 512
#define BN_EPS 1e-5f
#define MTILE 4                    // 4 x 16 = 64 edges per block

// ---------------------------------------------------------------------------
// Weight packing: emit B-matrix fragments in the CDNA5 WMMA 16x16x32 f16
// B layout: frag[((nt*KC + kc)*32 + lane)*16 + i] = W[nt*16 + (lane&15)]
//           [kc*32 + ((lane<16)?0:8) + (i<8 ? i : i+8)], zero-padded for k >= Ksrc.
// ---------------------------------------------------------------------------
__global__ void pack_b_frag(const float* __restrict__ src, half_t* __restrict__ dst,
                            int Ksrc, int KC, int NT) {
    long idx = (long)blockIdx.x * blockDim.x + threadIdx.x;
    long total = (long)NT * KC * 512;
    if (idx >= total) return;
    int i    = (int)(idx & 15);
    int lane = (int)((idx >> 4) & 31);
    int kc   = (int)((idx >> 9) % KC);
    int nt   = (int)(idx / ((long)KC * 512));
    int k = kc * 32 + ((lane < 16) ? 0 : 8) + ((i < 8) ? i : i + 8);
    int n = nt * 16 + (lane & 15);
    float v = (k < Ksrc) ? src[(size_t)n * Ksrc + k] : 0.0f;
    dst[idx] = (half_t)v;
}

// ---------------------------------------------------------------------------
// e = relu(edge_attr @ ee_w.T + ee_b)   [E,8] -> [E,16]  (f16 out)
// ---------------------------------------------------------------------------
__global__ void edge_embed_kernel(const float* __restrict__ ea,
                                  const float* __restrict__ w,
                                  const float* __restrict__ b,
                                  half_t* __restrict__ e16) {
    int idx = blockIdx.x * blockDim.x + threadIdx.x;
    if (idx >= N_EDGES * EHDIM) return;
    int e = idx >> 4, c = idx & 15;
    float acc = b[c];
#pragma unroll
    for (int k = 0; k < F_EDGE; ++k) acc += ea[(size_t)e * F_EDGE + k] * w[c * F_EDGE + k];
    e16[idx] = (half_t)(acc > 0.0f ? acc : 0.0f);
}

// ---------------------------------------------------------------------------
// h = relu(x @ ne_w.T + ne_b)   [N,32] -> [N,256]  (f16 out)
// ---------------------------------------------------------------------------
__global__ void node_embed_kernel(const float* __restrict__ x,
                                  const float* __restrict__ w,
                                  const float* __restrict__ b,
                                  half_t* __restrict__ h16) {
    int idx = blockIdx.x * blockDim.x + threadIdx.x;
    if (idx >= N_NODES * HDIM) return;
    int n = idx >> 8, c = idx & 255;
    float acc = b[c];
#pragma unroll
    for (int k = 0; k < F_NODE; ++k) acc += x[(size_t)n * F_NODE + k] * w[c * F_NODE + k];
    h16[idx] = (half_t)(acc > 0.0f ? acc : 0.0f);
}

__global__ void count_deg_kernel(const int* __restrict__ ei, int* __restrict__ cnt) {
    int e = blockIdx.x * blockDim.x + threadIdx.x;
    if (e >= N_EDGES) return;
    atomicAdd(&cnt[ei[N_EDGES + e]], 1);
}

// ---------------------------------------------------------------------------
// Extract an A fragment (16x32 f16) for one K-chunk from an LDS row-major
// tile. Per ISA 7.12.2: lane<16 -> K {0..7,16..23}, lane>=16 -> K {8..15,24..31}
// i.e. two contiguous 8-half (16B) reads at +sub and +sub+16.
// ---------------------------------------------------------------------------
__device__ __forceinline__ v16h make_a_frag(const half_t* base) {
    v8h lo = *(const v8h*)(base);
    v8h hi = *(const v8h*)(base + 16);
    v16h a;
#pragma unroll
    for (int i = 0; i < 8; ++i) { a[i] = lo[i]; a[8 + i] = hi[i]; }
    return a;
}

// ---------------------------------------------------------------------------
// Per-edge two-layer MLP with WMMA + scatter-add.  One 256-thread block
// (8 waves) per 64-edge tile (MTILE=4 M-subtiles); each B fragment loaded
// once per wave feeds 4 independent WMMAs (acc reuse distance 4 -> no D->A/B
// RAW hazard, 4x arithmetic intensity on the L2-resident weight stream).
// kc loops fully unrolled with 4 named accumulators so the allocator pins
// each accumulator to one 8-VGPR tuple (no v_mov/v_nop hazard traffic).
//   A  = [h[dst] | h[src] | e | 0pad]   (64 x 544 f16, LDS)
//   hid = relu(A @ W1^T + b1)           (64 x HID f16, LDS)
//   out = hid @ W2^T + b2               (64 x OUT f32) -> atomic sum[dst]
// ---------------------------------------------------------------------------
template <int HID, int OUT>
__global__ __launch_bounds__(256) void edge_mlp_kernel(
    const half_t* __restrict__ h16, const half_t* __restrict__ e16,
    const int* __restrict__ ei,
    const half_t* __restrict__ w1f, const float* __restrict__ b1,
    const half_t* __restrict__ w2f, const float* __restrict__ b2,
    float* __restrict__ sum) {
    constexpr int KC1 = 17;        // 544 / 32
    constexpr int NT1 = HID / 16;
    constexpr int KC2 = HID / 32;
    constexpr int NT2 = OUT / 16;
    constexpr int ROWS = MTILE * 16;

    extern __shared__ uint4 smem4[];
    half_t* Ash = (half_t*)smem4;                    // [ROWS][544]
    half_t* Hsh = Ash + (size_t)ROWS * 544;          // [ROWS][HID]
    __shared__ int sdst[ROWS];
    __shared__ int ssrc[ROWS];
    __shared__ int svalid[ROWS];

    const int tid  = threadIdx.x;
    const int lane = tid & 31;
    const int wid  = tid >> 5;
    const int mrow = lane & 15;
    const int sub  = (lane < 16) ? 0 : 8;
    const int ebase = blockIdx.x * ROWS;

    if (tid < ROWS) {
        int e = ebase + tid;
        int ok = (e < N_EDGES) ? 1 : 0;
        svalid[tid] = ok;
        ssrc[tid] = ok ? ei[e] : 0;
        sdst[tid] = ok ? ei[N_EDGES + e] : 0;
    }

    // Warm L2 with phase-2 weight fragments (global_prefetch_b8 path).
    if (blockIdx.x < 32) {
        const char* wp = (const char*)w2f;
        for (size_t off = (size_t)tid * 64; off < (size_t)NT2 * KC2 * 512 * 2; off += 256 * 64)
            __builtin_prefetch(wp + off, 0, 1);
    }
    __syncthreads();

    // Stage concatenated inputs: ROWS x 272 dwords (544 halves per row).
    {
        const unsigned* h32 = (const unsigned*)h16;
        const unsigned* e32 = (const unsigned*)e16;
        unsigned* A32 = (unsigned*)Ash;
        for (int idx = tid; idx < ROWS * 272; idx += 256) {
            int r = idx / 272, cd = idx - r * 272;
            unsigned v = 0u;
            if (svalid[r]) {
                if (cd < 128)      v = h32[(size_t)sdst[r] * 128 + cd];
                else if (cd < 256) v = h32[(size_t)ssrc[r] * 128 + (cd - 128)];
                else if (cd < 264) v = e32[(size_t)(ebase + r) * 8 + (cd - 256)];
            }
            A32[r * 272 + cd] = v;
        }
    }
    __syncthreads();

    const int mhi = (lane < 16) ? 0 : 8;

    // ---- Phase 1: hidden = relu(A @ W1^T + b1) ----
    {
        const half_t* a0 = Ash + (size_t)(0 * 16 + mrow) * 544 + sub;
        const half_t* a1 = Ash + (size_t)(1 * 16 + mrow) * 544 + sub;
        const half_t* a2 = Ash + (size_t)(2 * 16 + mrow) * 544 + sub;
        const half_t* a3 = Ash + (size_t)(3 * 16 + mrow) * 544 + sub;
        for (int nt = wid; nt < NT1; nt += 8) {
            const half_t* wrow = w1f + ((size_t)nt * KC1 * 32 + lane) * 16;
            v8f acc0 = {}, acc1 = {}, acc2 = {}, acc3 = {};
#pragma unroll
            for (int kc = 0; kc < KC1; ++kc) {
                v16h b = *(const v16h*)(wrow + (size_t)kc * 512);
                acc0 = __builtin_amdgcn_wmma_f32_16x16x32_f16(false, make_a_frag(a0 + kc * 32),
                                                              false, b, (short)0, acc0, false, false);
                acc1 = __builtin_amdgcn_wmma_f32_16x16x32_f16(false, make_a_frag(a1 + kc * 32),
                                                              false, b, (short)0, acc1, false, false);
                acc2 = __builtin_amdgcn_wmma_f32_16x16x32_f16(false, make_a_frag(a2 + kc * 32),
                                                              false, b, (short)0, acc2, false, false);
                acc3 = __builtin_amdgcn_wmma_f32_16x16x32_f16(false, make_a_frag(a3 + kc * 32),
                                                              false, b, (short)0, acc3, false, false);
            }
            int ncol = nt * 16 + mrow;
            float bias = b1[ncol];
            auto store_h = [&](v8f a, int m) {
#pragma unroll
                for (int v = 0; v < 8; ++v) {
                    float hv = a[v] + bias;
                    Hsh[(size_t)(m * 16 + v + mhi) * HID + ncol] = (half_t)(hv > 0.0f ? hv : 0.0f);
                }
            };
            store_h(acc0, 0); store_h(acc1, 1); store_h(acc2, 2); store_h(acc3, 3);
        }
    }
    __syncthreads();

    // ---- Phase 2: out = hidden @ W2^T + b2, scatter to sum[dst] ----
    {
        const half_t* a0 = Hsh + (size_t)(0 * 16 + mrow) * HID + sub;
        const half_t* a1 = Hsh + (size_t)(1 * 16 + mrow) * HID + sub;
        const half_t* a2 = Hsh + (size_t)(2 * 16 + mrow) * HID + sub;
        const half_t* a3 = Hsh + (size_t)(3 * 16 + mrow) * HID + sub;
        for (int nt = wid; nt < NT2; nt += 8) {
            const half_t* wrow = w2f + ((size_t)nt * KC2 * 32 + lane) * 16;
            v8f acc0 = {}, acc1 = {}, acc2 = {}, acc3 = {};
#pragma unroll
            for (int kc = 0; kc < KC2; ++kc) {
                v16h b = *(const v16h*)(wrow + (size_t)kc * 512);
                acc0 = __builtin_amdgcn_wmma_f32_16x16x32_f16(false, make_a_frag(a0 + kc * 32),
                                                              false, b, (short)0, acc0, false, false);
                acc1 = __builtin_amdgcn_wmma_f32_16x16x32_f16(false, make_a_frag(a1 + kc * 32),
                                                              false, b, (short)0, acc1, false, false);
                acc2 = __builtin_amdgcn_wmma_f32_16x16x32_f16(false, make_a_frag(a2 + kc * 32),
                                                              false, b, (short)0, acc2, false, false);
                acc3 = __builtin_amdgcn_wmma_f32_16x16x32_f16(false, make_a_frag(a3 + kc * 32),
                                                              false, b, (short)0, acc3, false, false);
            }
            int ncol = nt * 16 + mrow;
            float bias = b2[ncol];
            auto scatter = [&](v8f a, int m) {
#pragma unroll
                for (int v = 0; v < 8; ++v) {
                    int r = m * 16 + v + mhi;
                    if (svalid[r])
                        atomicAdd(&sum[(size_t)sdst[r] * OUT + ncol], a[v] + bias);
                }
            };
            scatter(acc0, 0); scatter(acc1, 1); scatter(acc2, 2); scatter(acc3, 3);
        }
    }
}

// ---------------------------------------------------------------------------
// h = relu(BN(sum / max(cnt,1)));  write f16 (next layer) and/or f32 (final)
// ---------------------------------------------------------------------------
__global__ void node_update_kernel(const float* __restrict__ sum, const int* __restrict__ cnt,
                                   const float* __restrict__ g, const float* __restrict__ bb,
                                   const float* __restrict__ rm, const float* __restrict__ rv,
                                   half_t* __restrict__ h16out, float* __restrict__ f32out,
                                   int C) {
    int idx = blockIdx.x * blockDim.x + threadIdx.x;
    if (idx >= N_NODES * C) return;
    int n = idx / C, c = idx - n * C;
    float cf = (float)cnt[n];
    if (cf < 1.0f) cf = 1.0f;
    float m = sum[idx] / cf;
    float y = (m - rm[c]) * rsqrtf(rv[c] + BN_EPS) * g[c] + bb[c];
    y = y > 0.0f ? y : 0.0f;
    if (h16out) h16out[idx] = (half_t)y;
    if (f32out) f32out[idx] = y;
}

__global__ void pool_accum_kernel(const float* __restrict__ hfin, const int* __restrict__ batch,
                                  float* __restrict__ psum, int* __restrict__ pcnt) {
    int idx = blockIdx.x * blockDim.x + threadIdx.x;
    if (idx >= N_NODES * 128) return;
    int n = idx >> 7, c = idx & 127;
    int grp = batch[n];
    atomicAdd(&psum[(size_t)grp * 128 + c], hfin[idx]);
    if (c == 0) atomicAdd(&pcnt[grp], 1);
}

__global__ void final_fc_kernel(const float* __restrict__ psum, const int* __restrict__ pcnt,
                                const float* __restrict__ fw, const float* __restrict__ fb,
                                float* __restrict__ out) {
    int idx = blockIdx.x * blockDim.x + threadIdx.x;
    if (idx >= GDIM * 12) return;
    int grp = idx / 12, o = idx - grp * 12;
    float cf = (float)pcnt[grp];
    if (cf < 1.0f) cf = 1.0f;
    float acc = fb[o];
    for (int c = 0; c < 128; ++c)
        acc += (psum[(size_t)grp * 128 + c] / cf) * fw[o * 128 + c];
    out[idx] = 1.0f / (1.0f + expf(-acc));
}

// ---------------------------------------------------------------------------
extern "C" void kernel_launch(void* const* d_in, const int* in_sizes, int n_in,
                              void* d_out, int out_size, void* d_ws, size_t ws_size,
                              hipStream_t stream) {
    (void)in_sizes; (void)n_in; (void)out_size; (void)ws_size;

    const float* x      = (const float*)d_in[0];
    const int*   ei     = (const int*)d_in[1];
    const float* eattr  = (const float*)d_in[2];
    const int*   batch  = (const int*)d_in[3];
    const float* ee_w   = (const float*)d_in[4];
    const float* ee_b   = (const float*)d_in[5];
    const float* ne_w   = (const float*)d_in[6];
    const float* ne_b   = (const float*)d_in[7];
    const float* w1[3]  = {(const float*)d_in[8],  (const float*)d_in[16], (const float*)d_in[24]};
    const float* b1[3]  = {(const float*)d_in[9],  (const float*)d_in[17], (const float*)d_in[25]};
    const float* w2[3]  = {(const float*)d_in[10], (const float*)d_in[18], (const float*)d_in[26]};
    const float* b2[3]  = {(const float*)d_in[11], (const float*)d_in[19], (const float*)d_in[27]};
    const float* bng[3] = {(const float*)d_in[12], (const float*)d_in[20], (const float*)d_in[28]};
    const float* bnb[3] = {(const float*)d_in[13], (const float*)d_in[21], (const float*)d_in[29]};
    const float* bnm[3] = {(const float*)d_in[14], (const float*)d_in[22], (const float*)d_in[30]};
    const float* bnv[3] = {(const float*)d_in[15], (const float*)d_in[23], (const float*)d_in[31]};
    const float* fc_w   = (const float*)d_in[32];
    const float* fc_b   = (const float*)d_in[33];
    float* out = (float*)d_out;

    // Layer dims: msg_in padded to 544 for all layers.
    const int HIDv[3] = {512, 512, 256};
    const int OUTv[3] = {256, 256, 128};

    // ---- Carve workspace (align 256B) ----
    uintptr_t p = (uintptr_t)d_ws;
    auto carve = [&](size_t bytes) -> void* {
        p = (p + 255) & ~(uintptr_t)255;
        void* r = (void*)p;
        p += bytes;
        return r;
    };
    half_t* e16    = (half_t*)carve((size_t)N_EDGES * EHDIM * 2);
    half_t* h16    = (half_t*)carve((size_t)N_NODES * HDIM * 2);
    float*  sum    = (float*)carve((size_t)N_NODES * HDIM * 4);
    int*    cnt    = (int*)carve((size_t)N_NODES * 4);
    float*  hfin   = (float*)carve((size_t)N_NODES * 128 * 4);
    float*  psum   = (float*)carve((size_t)GDIM * 128 * 4);
    int*    pcnt   = (int*)carve((size_t)GDIM * 4);
    half_t* w1f[3];
    half_t* w2f[3];
    for (int l = 0; l < 3; ++l) {
        w1f[l] = (half_t*)carve((size_t)(HIDv[l] / 16) * 17 * 512 * 2);
        w2f[l] = (half_t*)carve((size_t)(OUTv[l] / 16) * (HIDv[l] / 32) * 512 * 2);
    }

    // ---- Pack weights into WMMA B-fragment layout ----
    for (int l = 0; l < 3; ++l) {
        long t1 = (long)(HIDv[l] / 16) * 17 * 512;
        pack_b_frag<<<(int)((t1 + 255) / 256), 256, 0, stream>>>(w1[l], w1f[l], 528, 17, HIDv[l] / 16);
        long t2 = (long)(OUTv[l] / 16) * (HIDv[l] / 32) * 512;
        pack_b_frag<<<(int)((t2 + 255) / 256), 256, 0, stream>>>(w2[l], w2f[l], HIDv[l], HIDv[l] / 32, OUTv[l] / 16);
    }

    // ---- Embeddings + degree ----
    edge_embed_kernel<<<(N_EDGES * EHDIM + 255) / 256, 256, 0, stream>>>(eattr, ee_w, ee_b, e16);
    node_embed_kernel<<<(N_NODES * HDIM + 255) / 256, 256, 0, stream>>>(x, ne_w, ne_b, h16);
    hipMemsetAsync(cnt, 0, (size_t)N_NODES * 4, stream);
    count_deg_kernel<<<(N_EDGES + 255) / 256, 256, 0, stream>>>(ei, cnt);

    // ---- 3 conv layers ----
    const int rows_per_blk = MTILE * 16;
    const int tiles = (N_EDGES + rows_per_blk - 1) / rows_per_blk;
    for (int l = 0; l < 3; ++l) {
        hipMemsetAsync(sum, 0, (size_t)N_NODES * OUTv[l] * 4, stream);
        size_t lds = (size_t)rows_per_blk * (544 + HIDv[l]) * 2;
        if (l == 0 || l == 1) {
            edge_mlp_kernel<512, 256><<<tiles, 256, lds, stream>>>(
                h16, e16, ei, w1f[l], b1[l], w2f[l], b2[l], sum);
        } else {
            edge_mlp_kernel<256, 128><<<tiles, 256, lds, stream>>>(
                h16, e16, ei, w1f[l], b1[l], w2f[l], b2[l], sum);
        }
        if (l < 2) {
            node_update_kernel<<<(N_NODES * OUTv[l] + 255) / 256, 256, 0, stream>>>(
                sum, cnt, bng[l], bnb[l], bnm[l], bnv[l], h16, (float*)nullptr, OUTv[l]);
        } else {
            node_update_kernel<<<(N_NODES * OUTv[l] + 255) / 256, 256, 0, stream>>>(
                sum, cnt, bng[l], bnb[l], bnm[l], bnv[l], (half_t*)nullptr, hfin, OUTv[l]);
        }
    }

    // ---- Global mean pool + FC + sigmoid ----
    hipMemsetAsync(psum, 0, (size_t)GDIM * 128 * 4, stream);
    hipMemsetAsync(pcnt, 0, (size_t)GDIM * 4, stream);
    pool_accum_kernel<<<(N_NODES * 128 + 255) / 256, 256, 0, stream>>>(hfin, batch, psum, pcnt);
    final_fc_kernel<<<(GDIM * 12 + 255) / 256, 256, 0, stream>>>(psum, pcnt, fc_w, fc_b, out);
}
